// Net_17386027614281
// MI455X (gfx1250) — matrix operand order repaired and checked by reading, not compiled, not run on hardware
//
#include <hip/hip_runtime.h>

#define N_PAPER 200000
#define N_MESH  30000
#define N_NODES (N_PAPER + N_MESH)
#define D_PAPER 512
#define D_MESH  128
#define E_TRAIN 2000000
#define E_POS   500000
#define E_NEG   500000

typedef __attribute__((ext_vector_type(16))) _Float16 v16h;
typedef __attribute__((ext_vector_type(8)))  float    v8f;
typedef __attribute__((ext_vector_type(4)))  float    v4f;

static inline int cdiv(long a, long b) { return (int)((a + b - 1) / b); }

// ---------------------------------------------------------------------------
// Encode: out[row,0:16] = X[row,:] @ W + bias, via v_wmma_f32_16x16x32_f16.
// One wave per 16-row tile; K stepped by 32; X is streamed once (bigger than
// L2) so A loads are non-temporal b128 to keep the graph working set in L2.
// ---------------------------------------------------------------------------
__global__ __launch_bounds__(32) void encode_wmma(
    const float* __restrict__ X, const float* __restrict__ W,
    const float* __restrict__ bias, float* __restrict__ out,
    int D, int rowOffset)
{
  const int lane = threadIdx.x & 31;
  const int half = lane >> 4;     // 0: lanes 0-15, 1: lanes 16-31
  const int l15  = lane & 15;
  const int row0 = blockIdx.x * 16;
  const long arow = (long)(row0 + l15) * D;

  v8f c = {0.f, 0.f, 0.f, 0.f, 0.f, 0.f, 0.f, 0.f};
  for (int k = 0; k < D; k += 32) {
    v16h a, b;
    // A fragment (16x32 f16): lane<16 holds M=lane, K = k+0..7 (halves 0-7)
    // and K = k+16..23 (halves 8-15); lane>=16 holds K = k+8..15 / k+24..31.
    const v4f* ap = (const v4f*)(X + arow + k + (half ? 8 : 0));
    const v4f f0 = __builtin_nontemporal_load(ap + 0);  // K+0..3
    const v4f f1 = __builtin_nontemporal_load(ap + 1);  // K+4..7
    const v4f f2 = __builtin_nontemporal_load(ap + 4);  // K+16..19
    const v4f f3 = __builtin_nontemporal_load(ap + 5);  // K+20..23
#pragma unroll
    for (int i = 0; i < 4; ++i) {
      a[i]      = (_Float16)f0[i];
      a[4 + i]  = (_Float16)f1[i];
      a[8 + i]  = (_Float16)f2[i];
      a[12 + i] = (_Float16)f3[i];
    }
    // B fragment (32x16 f16): lane<16 holds N=lane, K = k+0..7 / k+16..23;
    // lane>=16 holds N=lane-16, K = k+8..15 / k+24..31.  W is tiny -> cache-hot.
    const int kb = k + (half ? 8 : 0);
#pragma unroll
    for (int i = 0; i < 16; ++i)
      b[i] = (_Float16)W[(kb + i + (i >= 8 ? 8 : 0)) * 16 + l15];

    c = __builtin_amdgcn_wmma_f32_16x16x32_f16(
        /*neg_a=*/false, a, /*neg_b=*/false, b,
        /*c_mod=*/(short)0, c, /*reuse_a=*/false, /*reuse_b=*/false);
  }
  // C layout: VGPR v -> row v (lanes 0-15) or row v+8 (lanes 16-31), col l15.
  const float bv = bias[l15];
#pragma unroll
  for (int v = 0; v < 8; ++v) {
    int r = row0 + v + (half ? 8 : 0);
    out[(long)(rowOffset + r) * 16 + l15] = c[v] + bv;
  }
}

// ---------------------------------------------------------------------------
// W_r = sum_b comp[r,b] * basis[b]   (tiny: <= 2*32*32 floats)
// ---------------------------------------------------------------------------
__global__ void make_weights(const float* __restrict__ comp,
                             const float* __restrict__ basis,
                             float* __restrict__ Wout, int IN, int OUT)
{
  int idx = blockIdx.x * blockDim.x + threadIdx.x;
  int total = 2 * IN * OUT;
  if (idx >= total) return;
  int r  = idx / (IN * OUT);
  int io = idx - r * (IN * OUT);
  float s = 0.f;
#pragma unroll
  for (int b = 0; b < 4; ++b)
    s += comp[r * 4 + b] * basis[b * IN * OUT + io];
  Wout[idx] = s;
}

// ---------------------------------------------------------------------------
// deg[dst] += 1 over the training edges (same dst set for all 3 layers).
// ---------------------------------------------------------------------------
__global__ void degree_kernel(const int* __restrict__ dst,
                              float* __restrict__ deg, int E)
{
  int e = blockIdx.x * blockDim.x + threadIdx.x;
  if (e < E) atomicAdd(&deg[dst[e]], 1.0f);
}

// ---------------------------------------------------------------------------
// H[r][n][o] = x[n,:] @ W_r[:,o] for r = 0,1, via WMMA.
// One wave per 16-node tile; single K step (IN<=32, zero-padded to 32);
// (2*OUT)/16 B-tiles over the concatenated [W_0 | W_1] columns.
// ---------------------------------------------------------------------------
template <int IN, int OUT>
__global__ __launch_bounds__(32) void precompute_h_wmma(
    const float* __restrict__ x, const float* __restrict__ Wrio, // [2,IN,OUT]
    float* __restrict__ H, int Nn)
{
  const int lane = threadIdx.x & 31;
  const int half = lane >> 4;
  const int l15  = lane & 15;
  const int row0 = blockIdx.x * 16;

  // A fragment: 16 nodes x K=32 (zero-padded when IN==16).
  v16h a;
  const float* xr = x + (long)(row0 + l15) * IN;
#pragma unroll
  for (int i = 0; i < 8; ++i) {
    int k0 = i + (half ? 8 : 0);        // halves 0..7  -> K 0..7  / 8..15
    int k1 = 16 + k0;                   // halves 8..15 -> K 16..23 / 24..31
    a[i]     = (k0 < IN) ? (_Float16)xr[k0] : (_Float16)0.f;
    a[8 + i] = (k1 < IN) ? (_Float16)xr[k1] : (_Float16)0.f;
  }

  const int NT = (2 * OUT) / 16;        // B column tiles over [W_0 | W_1]
#pragma unroll
  for (int t = 0; t < NT; ++t) {
    const int col = t * 16 + l15;       // column in [0, 2*OUT)
    const int r   = col / OUT;
    const int o   = col - r * OUT;
    const float* wcol = Wrio + (long)r * IN * OUT + o;
    v16h b;
#pragma unroll
    for (int i = 0; i < 16; ++i) {
      int kk = i + (half ? 8 : 0) + (i >= 8 ? 8 : 0);
      b[i] = (kk < IN) ? (_Float16)wcol[(long)kk * OUT] : (_Float16)0.f;
    }
    v8f c = {0.f, 0.f, 0.f, 0.f, 0.f, 0.f, 0.f, 0.f};
    c = __builtin_amdgcn_wmma_f32_16x16x32_f16(
        false, a, false, b, (short)0, c, false, false);
    float* Hb = H + (long)r * Nn * OUT;
#pragma unroll
    for (int v = 0; v < 8; ++v) {
      int n = row0 + v + (half ? 8 : 0);
      Hb[(long)n * OUT + o] = c[v];
    }
  }
}

// ---------------------------------------------------------------------------
// Scatter-mean numerator: agg[dst] += H[etype][src].  OUT/4 threads per edge,
// each moves a float4 (8 lanes cover a full 128B row for OUT=32) so gathers
// stay cacheline-coherent; atomics resolve in L2 (all buffers L2-resident).
// ---------------------------------------------------------------------------
template <int OUT>
__global__ void scatter_kernel(const int* __restrict__ ei,   // [2,E]
                               const int* __restrict__ et,
                               const float* __restrict__ H,  // [2,N,OUT]
                               float* __restrict__ agg, int E, int Nn)
{
  const int TPE = OUT / 4;
  int gid = blockIdx.x * blockDim.x + threadIdx.x;
  if (gid >= E * TPE) return;
  int e = gid / TPE;
  int c = (gid - e * TPE) * 4;
  int s = ei[e];
  int d = ei[E + e];
  int r = et[e] ? 1 : 0;
  const v4f v = *(const v4f*)(H + ((long)r * Nn + s) * OUT + c);
  float* ap = agg + (long)d * OUT + c;
  atomicAdd(ap + 0, v[0]);
  atomicAdd(ap + 1, v[1]);
  atomicAdd(ap + 2, v[2]);
  atomicAdd(ap + 3, v[3]);
}

// ---------------------------------------------------------------------------
// out[n][o] = agg[n][o]/max(deg[n],1) + x[n,:]@root[:,o] + bias[o]  (+ReLU)
// ---------------------------------------------------------------------------
template <int IN, int OUT, bool RELU>
__global__ void combine_kernel(const float* __restrict__ x,
                               const float* __restrict__ agg,
                               const float* __restrict__ deg,
                               const float* __restrict__ root,  // [IN,OUT]
                               const float* __restrict__ bias,
                               float* __restrict__ out, int Nn)
{
  __shared__ float sW[IN * OUT];
  __shared__ float sB[OUT];
  for (int i = threadIdx.x; i < IN * OUT; i += blockDim.x) sW[i] = root[i];
  for (int i = threadIdx.x; i < OUT; i += blockDim.x) sB[i] = bias[i];
  __syncthreads();
  int gid = blockIdx.x * blockDim.x + threadIdx.x;
  if (gid >= Nn * OUT) return;
  int n = gid / OUT;
  int o = gid - n * OUT;
  const float* xr = x + (long)n * IN;
  float acc = 0.f;
#pragma unroll
  for (int i = 0; i < IN; ++i) acc += xr[i] * sW[i * OUT + o];
  float dg = deg[n];
  dg = dg < 1.f ? 1.f : dg;
  float v = agg[gid] / dg + acc + sB[o];
  out[gid] = (RELU && v < 0.f) ? 0.f : v;
}

// ---------------------------------------------------------------------------
// Decode: score = ((z[src]@W_t + b_t) * z[dst]).sum(), W_t selected per edge.
// Both 16x16 weight sets + biases live in LDS.
// ---------------------------------------------------------------------------
__global__ void decode_kernel(const int* __restrict__ ei,  // [2,E]
                              const int* __restrict__ et,
                              const float* __restrict__ z, // [N,16]
                              const float* __restrict__ dpp_w,
                              const float* __restrict__ dpp_b,
                              const float* __restrict__ dpm_w,
                              const float* __restrict__ dpm_b,
                              float* __restrict__ out, int E)
{
  __shared__ float sw[2][16][16];
  __shared__ float sb[2][16];
  for (int i = threadIdx.x; i < 256; i += blockDim.x) {
    sw[0][i / 16][i % 16] = dpp_w[i];
    sw[1][i / 16][i % 16] = dpm_w[i];
  }
  for (int i = threadIdx.x; i < 16; i += blockDim.x) {
    sb[0][i] = dpp_b[i];
    sb[1][i] = dpm_b[i];
  }
  __syncthreads();
  int e = blockIdx.x * blockDim.x + threadIdx.x;
  if (e >= E) return;
  int s = ei[e];
  int d = ei[E + e];
  int r = et[e] ? 1 : 0;
  float zs[16], zd[16];
  const float* zsp = z + (long)s * 16;
  const float* zdp = z + (long)d * 16;
#pragma unroll
  for (int i = 0; i < 16; ++i) { zs[i] = zsp[i]; zd[i] = zdp[i]; }
  float acc = 0.f;
#pragma unroll
  for (int j = 0; j < 16; ++j) {
    float p = sb[r][j];
#pragma unroll
    for (int i = 0; i < 16; ++i) p += zs[i] * sw[r][i][j];
    acc += p * zd[j];
  }
  out[e] = acc;
}

// ---------------------------------------------------------------------------
extern "C" void kernel_launch(void* const* d_in, const int* in_sizes, int n_in,
                              void* d_out, int out_size, void* d_ws, size_t ws_size,
                              hipStream_t stream)
{
  const float* x_paper = (const float*)d_in[0];
  const float* x_mesh  = (const float*)d_in[1];
  const float* tp_w = (const float*)d_in[2];
  const float* tp_b = (const float*)d_in[3];
  const float* tm_w = (const float*)d_in[4];
  const float* tm_b = (const float*)d_in[5];
  const float* comp1  = (const float*)d_in[6];
  const float* basis1 = (const float*)d_in[7];
  const float* root1  = (const float*)d_in[8];
  const float* bias1  = (const float*)d_in[9];
  const float* comp2  = (const float*)d_in[10];
  const float* basis2 = (const float*)d_in[11];
  const float* root2  = (const float*)d_in[12];
  const float* bias2  = (const float*)d_in[13];
  const float* comp3  = (const float*)d_in[14];
  const float* basis3 = (const float*)d_in[15];
  const float* root3  = (const float*)d_in[16];
  const float* bias3  = (const float*)d_in[17];
  const float* dpp_w  = (const float*)d_in[18];
  const float* dpp_b  = (const float*)d_in[19];
  const float* dpm_w  = (const float*)d_in[20];
  const float* dpm_b  = (const float*)d_in[21];
  const int* train_ei = (const int*)d_in[22];
  const int* train_et = (const int*)d_in[23];
  const int* pos_ei   = (const int*)d_in[24];
  const int* pos_et   = (const int*)d_in[25];
  const int* neg_ei   = (const int*)d_in[26];
  const int* neg_et   = (const int*)d_in[27];
  float* out = (float*)d_out;

  // Workspace layout (floats). z aliases x0 (x0 is dead after layer 1).
  float* ws = (float*)d_ws;
  size_t off = 0;
  float* x0  = ws + off; off += (size_t)N_NODES * 16;   // encode out / z
  float* x1  = ws + off; off += (size_t)N_NODES * 32;
  float* x2  = ws + off; off += (size_t)N_NODES * 32;
  float* H   = ws + off; off += (size_t)N_NODES * 64;   // H[0] | H[1]
  float* agg = ws + off; off += (size_t)N_NODES * 32;
  float* deg = ws + off; off += (size_t)N_NODES;
  float* Wl  = ws + off; off += 2048;                   // per-layer W_r
  float* z   = x0;

  const int B = 256;

  // ---- encode (WMMA, NT streaming loads) ----
  encode_wmma<<<N_PAPER / 16, 32, 0, stream>>>(x_paper, tp_w, tp_b, x0, D_PAPER, 0);
  encode_wmma<<<N_MESH / 16, 32, 0, stream>>>(x_mesh, tm_w, tm_b, x0, D_MESH, N_PAPER);

  // ---- degrees (shared across all three layers) ----
  (void)hipMemsetAsync(deg, 0, (size_t)N_NODES * sizeof(float), stream);
  degree_kernel<<<cdiv(E_TRAIN, B), B, 0, stream>>>(train_ei + E_TRAIN, deg, E_TRAIN);

  // ---- layer 1: 16 -> 32, ReLU ----
  make_weights<<<cdiv(2 * 16 * 32, B), B, 0, stream>>>(comp1, basis1, Wl, 16, 32);
  precompute_h_wmma<16, 32><<<N_NODES / 16, 32, 0, stream>>>(x0, Wl, H, N_NODES);
  (void)hipMemsetAsync(agg, 0, (size_t)N_NODES * 32 * sizeof(float), stream);
  scatter_kernel<32><<<cdiv((long)E_TRAIN * 8, B), B, 0, stream>>>(train_ei, train_et, H, agg, E_TRAIN, N_NODES);
  combine_kernel<16, 32, true><<<cdiv((long)N_NODES * 32, B), B, 0, stream>>>(x0, agg, deg, root1, bias1, x1, N_NODES);

  // ---- layer 2: 32 -> 32, ReLU ----
  make_weights<<<cdiv(2 * 32 * 32, B), B, 0, stream>>>(comp2, basis2, Wl, 32, 32);
  precompute_h_wmma<32, 32><<<N_NODES / 16, 32, 0, stream>>>(x1, Wl, H, N_NODES);
  (void)hipMemsetAsync(agg, 0, (size_t)N_NODES * 32 * sizeof(float), stream);
  scatter_kernel<32><<<cdiv((long)E_TRAIN * 8, B), B, 0, stream>>>(train_ei, train_et, H, agg, E_TRAIN, N_NODES);
  combine_kernel<32, 32, true><<<cdiv((long)N_NODES * 32, B), B, 0, stream>>>(x1, agg, deg, root2, bias2, x2, N_NODES);

  // ---- layer 3: 32 -> 16, no ReLU ----
  make_weights<<<cdiv(2 * 32 * 16, B), B, 0, stream>>>(comp3, basis3, Wl, 32, 16);
  precompute_h_wmma<32, 16><<<N_NODES / 16, 32, 0, stream>>>(x2, Wl, H, N_NODES);
  (void)hipMemsetAsync(agg, 0, (size_t)N_NODES * 16 * sizeof(float), stream);
  scatter_kernel<16><<<cdiv((long)E_TRAIN * 4, B), B, 0, stream>>>(train_ei, train_et, H, agg, E_TRAIN, N_NODES);
  combine_kernel<32, 16, false><<<cdiv((long)N_NODES * 16, B), B, 0, stream>>>(x2, agg, deg, root3, bias3, z, N_NODES);

  // ---- decode: pos then neg, concatenated ----
  decode_kernel<<<cdiv(E_POS, B), B, 0, stream>>>(pos_ei, pos_et, z, dpp_w, dpp_b, dpm_w, dpm_b, out, E_POS);
  decode_kernel<<<cdiv(E_NEG, B), B, 0, stream>>>(neg_ei, neg_et, z, dpp_w, dpp_b, dpm_w, dpm_b, out + E_POS, E_NEG);
}
